// CustomDense_36283883717254
// MI455X (gfx1250) — compile-verified, matrix operand order checked by reading, not run on hardware
//
#include <hip/hip_runtime.h>

// ---- CDNA5 / gfx1250 bf16 WMMA GEMM:  C[M,N] = A[M,K] * W[N,K]^T ----
// M=131072, K=256, N=256, fp32 in/out, bf16 WMMA compute (memory-bound design).
// Round 2 fix: explicit 4-deep rotating B-fragment buffer (prefetch distance 2)
// so each WMMA waits on dscnt<=4 instead of a full DScnt drain.

typedef __attribute__((ext_vector_type(16))) __bf16       v16bf;
typedef __attribute__((ext_vector_type(8)))  __bf16       v8bf;
typedef __attribute__((ext_vector_type(8)))  float        v8f;
typedef __attribute__((ext_vector_type(4)))  float        v4f;
typedef __attribute__((ext_vector_type(4)))  unsigned int v4u;

#define M_TOK 131072
#define K_DIM 256
#define N_DIM 256
#define BLOCK_THREADS 256          // 8 wave32
#define WAVES 8
#define MT_PER_WAVE 2              // M-tiles (16 rows) per wave, sequential
#define BLOCK_ROWS (WAVES * MT_PER_WAVE * 16)   // 256 rows per block
#define NT_TILES (N_DIM / 16)      // 16
#define KC_CHUNKS (K_DIM / 32)     // 8
#define N_FRAGS (KC_CHUNKS * NT_TILES)          // 128 B-fragments
#define LDS_FRAGS (N_FRAGS + 2)                 // +2 pad slots for tail prefetch

struct BFragRaw { v4u lo, hi; };   // 32B: one per-lane B fragment (16 bf16)

__device__ __forceinline__ v16bf cvt_a_frag(v4f a0, v4f a1, v4f a2, v4f a3) {
    v16bf a;
    a[0]  = (__bf16)a0.x; a[1]  = (__bf16)a0.y; a[2]  = (__bf16)a0.z; a[3]  = (__bf16)a0.w;
    a[4]  = (__bf16)a1.x; a[5]  = (__bf16)a1.y; a[6]  = (__bf16)a1.z; a[7]  = (__bf16)a1.w;
    a[8]  = (__bf16)a2.x; a[9]  = (__bf16)a2.y; a[10] = (__bf16)a2.z; a[11] = (__bf16)a2.w;
    a[12] = (__bf16)a3.x; a[13] = (__bf16)a3.y; a[14] = (__bf16)a3.z; a[15] = (__bf16)a3.w;
    return a;
}

__device__ __forceinline__ v16bf load_b_frag(const v4u* ldsB, int f, int lane) {
    BFragRaw r { ldsB[f * 64 + lane], ldsB[f * 64 + 32 + lane] };
    return __builtin_bit_cast(v16bf, r);
}

__global__ __launch_bounds__(BLOCK_THREADS)
void CustomDense_gemm_bf16_wmma(const float* __restrict__ A,
                                const float* __restrict__ W,
                                float* __restrict__ C)
{
    // B fragments for the whole weight: 128 frags * 64 v4u = 128 KB (+2 KB pad)
    __shared__ v4u ldsB[LDS_FRAGS * 64];

    const int t    = threadIdx.x;
    const int lane = t & 31;
    const int wave = t >> 5;

    // ---- Stage W (f32, [N,K] row-major) into LDS as bf16 WMMA B-fragments ----
    // Fragment layout per lane (mirror of 16-bit A 16x32, ISA 7.12.2):
    //   lane l (N = nt*16 + (l&15)), kh = l>=16:
    //     elements 0..7  -> K = kc*32 + kh*8      + {0..7}
    //     elements 8..15 -> K = kc*32 + 16 + kh*8 + {0..7}
    #pragma unroll
    for (int i = 0; i < 32; ++i) {
        int u  = i * BLOCK_THREADS + t;    // 0..8191
        int n  = u >> 5;                   // weight row = output column
        int r  = u & 31;                   // 8-float run within row
        int kh = r & 1;
        int h  = (r >> 1) & 1;
        int kc = r >> 2;
        int k0 = kc * 32 + h * 16 + kh * 8;
        const v4f* src = reinterpret_cast<const v4f*>(W + n * K_DIM + k0);
        v4f f0 = src[0];
        v4f f1 = src[1];
        v8bf b;
        b[0] = (__bf16)f0.x; b[1] = (__bf16)f0.y; b[2] = (__bf16)f0.z; b[3] = (__bf16)f0.w;
        b[4] = (__bf16)f1.x; b[5] = (__bf16)f1.y; b[6] = (__bf16)f1.z; b[7] = (__bf16)f1.w;
        int nt  = n >> 4;
        int l   = (n & 15) | (kh << 4);
        int dst = ((kc * NT_TILES + nt) * 2 + h) * 32 + l;
        ldsB[dst] = __builtin_bit_cast(v4u, b);
    }
    __syncthreads();

    const int blockRow = blockIdx.x * BLOCK_ROWS;

    #pragma unroll 1
    for (int mt = 0; mt < MT_PER_WAVE; ++mt) {
        const int rowBase = blockRow + wave * (MT_PER_WAVE * 16) + mt * 16;

        // A fragment addressing (16-bit A 16x32 layout): lane holds row rowBase+(l&15);
        // upper half-wave takes K runs offset by +8.
        const int    aRow  = rowBase + (lane & 15);
        const int    khalf = (lane >> 4) * 8;
        const float* aPtr  = A + (size_t)aRow * K_DIM + khalf;

        // Prefetch the next M-tile's A rows while this tile computes.
        if (mt + 1 < MT_PER_WAVE)
            __builtin_prefetch(aPtr + 16 * K_DIM, 0, 0);   // global_prefetch_b8

        v8f acc[NT_TILES];
        #pragma unroll
        for (int nt = 0; nt < NT_TILES; ++nt)
            acc[nt] = (v8f){0.f, 0.f, 0.f, 0.f, 0.f, 0.f, 0.f, 0.f};

        // Software-pipelined A loads: issue kc+1's loads before kc's WMMA burst.
        v4f a0 = *reinterpret_cast<const v4f*>(aPtr);
        v4f a1 = *reinterpret_cast<const v4f*>(aPtr + 4);
        v4f a2 = *reinterpret_cast<const v4f*>(aPtr + 16);
        v4f a3 = *reinterpret_cast<const v4f*>(aPtr + 20);

        // Prime the rotating B-fragment pipeline (depth 2, 4 buffers; 16 % 4 == 0
        // keeps the rotation phase consistent across kc iterations).
        v16bf bbuf[4];
        bbuf[0] = load_b_frag(ldsB, 0, lane);
        bbuf[1] = load_b_frag(ldsB, 1, lane);

        #pragma unroll 1
        for (int kc = 0; kc < KC_CHUNKS; ++kc) {
            v16bf a = cvt_a_frag(a0, a1, a2, a3);

            if (kc + 1 < KC_CHUNKS) {
                const float* ap = aPtr + (kc + 1) * 32;
                a0 = *reinterpret_cast<const v4f*>(ap);
                a1 = *reinterpret_cast<const v4f*>(ap + 4);
                a2 = *reinterpret_cast<const v4f*>(ap + 16);
                a3 = *reinterpret_cast<const v4f*>(ap + 20);
            }

            #pragma unroll
            for (int nt = 0; nt < NT_TILES; ++nt) {
                // Prefetch fragment (kc,nt+2); contiguous layout rolls into kc+1,
                // and the 2 pad slots absorb the tail of the last kc.
                bbuf[(nt + 2) & 3] =
                    load_b_frag(ldsB, kc * NT_TILES + nt + 2, lane);
                // (neg_a, A, neg_b, B, c_mod, C, reuse_a, reuse_b)
                acc[nt] = __builtin_amdgcn_wmma_f32_16x16x32_bf16(
                    false, a, false, bbuf[nt & 3], (short)0, acc[nt],
                    false, false);
            }
        }

        // ---- Store C: acc VGPR j -> row rowBase + j + (lane>=16 ? 8 : 0), col nt*16+(lane&15)
        const int cRow0 = rowBase + (lane >> 4) * 8;
        const int cCol  = lane & 15;
        #pragma unroll
        for (int nt = 0; nt < NT_TILES; ++nt) {
            #pragma unroll
            for (int j = 0; j < 8; ++j) {
                C[(size_t)(cRow0 + j) * N_DIM + nt * 16 + cCol] = acc[nt][j];
            }
        }
    }
}

extern "C" void kernel_launch(void* const* d_in, const int* in_sizes, int n_in,
                              void* d_out, int out_size, void* d_ws, size_t ws_size,
                              hipStream_t stream) {
    (void)in_sizes; (void)n_in; (void)d_ws; (void)ws_size; (void)out_size;
    const float* A = (const float*)d_in[0];   // input  [M,K] f32
    const float* W = (const float*)d_in[1];   // weight [N,K] f32
    // d_in[2] = approximate (kernel flag; constant w.r.t. the GEMM) -- unused
    float* C = (float*)d_out;                 // output [M,N] f32

    dim3 grid(M_TOK / BLOCK_ROWS);            // 512 blocks
    dim3 block(BLOCK_THREADS);                // 256 threads = 8 wave32
    CustomDense_gemm_bf16_wmma<<<grid, block, 0, stream>>>(A, W, C);
}
